// Decoder_36953898615460
// MI455X (gfx1250) — matrix operand order repaired and checked by reading, not compiled, not run on hardware
//
#include <hip/hip_runtime.h>

typedef __attribute__((ext_vector_type(16))) __bf16 v16bf;
typedef __attribute__((ext_vector_type(2)))  __bf16 v2bf;
typedef __attribute__((ext_vector_type(8)))  float  v8f;

#define B_DIM   2048
#define F_DIM   6144
#define D_DIM   768
#define BM      128
#define BN      128
#define BK      64
#define LDA     72              // BK + 8 pad (bf16 elems) -> every row 16B aligned
#define KSTAGES (F_DIM / BK)    // 96 stages per layer
#define SMEM_BYTES (2 * BM * LDA * 2 + 2 * BN * LDA * 2 + BN * 4)   // 74240

typedef unsigned int   u32;
typedef unsigned short u16;

struct alignas(16) U4 { u32 x, y, z, w; };
union FragU { U4 q[2]; v16bf v; };

// f32 pair -> packed bf16 dword. Plain casts: backend selects v_cvt_pk_bf16_f32
// if the target has it, else LLVM's own RNE expansion.
__device__ __forceinline__ u32 pack2_bf16(float lo, float hi) {
    v2bf v;
    v[0] = (__bf16)lo;
    v[1] = (__bf16)hi;
    return __builtin_bit_cast(u32, v);
}

__global__ __launch_bounds__(256) void decoder_gemm_wmma(
    const float* __restrict__ acts,   // [L,B,F]
    const float* __restrict__ W,      // [L,F,D]
    const float* __restrict__ bias,   // [L,D]
    const int*   __restrict__ p_layer,
    float*       __restrict__ out)    // [B,D]
{
    extern __shared__ char smem[];
    u16*   shA    = (u16*)smem;                    // 2 x [BM][LDA] bf16, row-major [m][k]
    u16*   shB    = shA + 2 * BM * LDA;            // 2 x [BN][LDA] bf16, TRANSPOSED [d][k]
    float* shBias = (float*)(shB + 2 * BN * LDA);  // [BN]

    const int tid = threadIdx.x;
    const int bx  = blockIdx.x;          // D tile: 0..5
    const int by  = blockIdx.y;          // B tile: 0..15
    const int n_layers = p_layer[0] + 1;

    // per-column bias sums (first loop barrier covers the hazard)
    if (tid < BN) {
        float s = 0.f;
        for (int l = 0; l < n_layers; ++l) s += bias[(size_t)l * D_DIM + bx * BN + tid];
        shBias[tid] = s;
    }

    const int lane   = tid & 31;
    const int wid    = tid >> 5;          // 0..7
    const int wm     = (wid >> 2) * 64;   // wave sub-tile: 64 rows
    const int wn     = (wid & 3)  * 32;   // wave sub-tile: 32 cols
    const int laneLo = lane & 15;
    const int laneHi = (lane >> 4) & 1;

    const v8f vzero = {0.f, 0.f, 0.f, 0.f, 0.f, 0.f, 0.f, 0.f};
    v8f acc[4][2];
#pragma unroll
    for (int i = 0; i < 4; ++i)
#pragma unroll
        for (int j = 0; j < 2; ++j) acc[i][j] = vzero;

    const int nStages = n_layers * KSTAGES;

    float4 aR[8];   // staged A f32: 128x64 / 256 thr = 32 floats
    float4 wR[8];   // staged W f32: 64x128 / 256 thr = 32 floats

    auto loadStage = [&](int s) {
        const int l  = s / KSTAGES;
        const int k0 = (s - l * KSTAGES) * BK;
        const float* Ag = acts + (size_t)l * B_DIM * F_DIM + (size_t)(by * BM) * F_DIM + k0;
        const float* Wg = W    + (size_t)l * F_DIM * D_DIM + (size_t)k0 * D_DIM + bx * BN;
#pragma unroll
        for (int i = 0; i < 4; ++i) {
            const int chunk = tid + i * 256;          // 0..1023, 8 floats each
            const int row   = chunk >> 3;             // 8 chunks per 64-float row
            const int col   = (chunk & 7) << 3;
            const float4* p = (const float4*)(Ag + (size_t)row * F_DIM + col);
            aR[2 * i]     = p[0];
            aR[2 * i + 1] = p[1];
        }
#pragma unroll
        for (int i = 0; i < 4; ++i) {
            const int chunk = tid + i * 256;          // 0..1023
            const int pr    = chunk >> 5;             // pair of f-rows, 0..31
            const int dc    = (chunk & 31) << 2;      // 4 consecutive d
            const float* p  = Wg + (size_t)(2 * pr) * D_DIM + dc;
            wR[2 * i]     = *(const float4*)p;
            wR[2 * i + 1] = *(const float4*)(p + D_DIM);
        }
    };

    auto storeStage = [&](int buf) {
        u16* bufA = shA + buf * (BM * LDA);
        u16* bufB = shB + buf * (BN * LDA);
#pragma unroll
        for (int i = 0; i < 4; ++i) {
            const int chunk = tid + i * 256;
            const int row   = chunk >> 3;
            const int col   = (chunk & 7) << 3;
            U4 v;
            v.x = pack2_bf16(aR[2 * i].x,     aR[2 * i].y);
            v.y = pack2_bf16(aR[2 * i].z,     aR[2 * i].w);
            v.z = pack2_bf16(aR[2 * i + 1].x, aR[2 * i + 1].y);
            v.w = pack2_bf16(aR[2 * i + 1].z, aR[2 * i + 1].w);
            *(U4*)&bufA[row * LDA + col] = v;
        }
#pragma unroll
        for (int i = 0; i < 4; ++i) {
            const int chunk = tid + i * 256;
            const int pr    = chunk >> 5;
            const int dc    = (chunk & 31) << 2;
            const int kk    = 2 * pr;                 // even k -> dword aligned
            *(u32*)&bufB[(dc + 0) * LDA + kk] = pack2_bf16(wR[2 * i].x, wR[2 * i + 1].x);
            *(u32*)&bufB[(dc + 1) * LDA + kk] = pack2_bf16(wR[2 * i].y, wR[2 * i + 1].y);
            *(u32*)&bufB[(dc + 2) * LDA + kk] = pack2_bf16(wR[2 * i].z, wR[2 * i + 1].z);
            *(u32*)&bufB[(dc + 3) * LDA + kk] = pack2_bf16(wR[2 * i].w, wR[2 * i + 1].w);
        }
    };

    // prologue: stage 0 into buffer 0
    loadStage(0);
    storeStage(0);

    for (int s = 0; s < nStages; ++s) {
        __syncthreads();                       // buf(s) ready; buf(s^1) readers done
        const int buf = s & 1;
        if (s + 1 < nStages) loadStage(s + 1); // issue next-stage HBM loads early

        const u16* bufA = shA + buf * (BM * LDA);
        const u16* bufB = shB + buf * (BN * LDA);

#pragma unroll
        for (int ks = 0; ks < BK; ks += 32) {
            // ---- fragments from LDS (layouts per CDNA5 ISA 7.12.2) ----
            FragU a[4];
#pragma unroll
            for (int mi = 0; mi < 4; ++mi) {
                const u16* pa = &bufA[(wm + mi * 16 + laneLo) * LDA + ks + laneHi * 8];
                a[mi].q[0] = *(const U4*)pa;          // K 0..7   (or 8..15)
                a[mi].q[1] = *(const U4*)(pa + 16);   // K 16..23 (or 24..31)
            }
            FragU b[2];
#pragma unroll
            for (int ni = 0; ni < 2; ++ni) {
                const u16* pb = &bufB[(wn + ni * 16 + laneLo) * LDA + ks + laneHi * 8];
                b[ni].q[0] = *(const U4*)pb;          // K 0..7   (or 8..15)
                b[ni].q[1] = *(const U4*)(pb + 16);   // K 16..23 (or 24..31)
            }
            // ---- 8 matrix ops per k-step per wave ----
#pragma unroll
            for (int mi = 0; mi < 4; ++mi)
#pragma unroll
                for (int ni = 0; ni < 2; ++ni)
                    acc[mi][ni] = __builtin_amdgcn_wmma_f32_16x16x32_bf16(
                        false, a[mi].v, false, b[ni].v,
                        (short)0, acc[mi][ni], false, false);
        }

        if (s + 1 < nStages) storeStage(buf ^ 1);  // convert + fill other buffer
    }

    // ---- epilogue: C layout = VGPR r -> M=r (lanes 0-15) / M=r+8 (lanes 16-31) ----
    const int gm0 = by * BM + wm + laneHi * 8;
    const int gn0 = bx * BN + wn + laneLo;
#pragma unroll
    for (int mi = 0; mi < 4; ++mi) {
#pragma unroll
        for (int ni = 0; ni < 2; ++ni) {
            const int   gn = gn0 + ni * 16;
            const float bs = shBias[wn + ni * 16 + laneLo];
#pragma unroll
            for (int r = 0; r < 8; ++r)
                out[(size_t)(gm0 + mi * 16 + r) * D_DIM + gn] = acc[mi][ni][r] + bs;
        }
    }
}

extern "C" void kernel_launch(void* const* d_in, const int* in_sizes, int n_in,
                              void* d_out, int out_size, void* d_ws, size_t ws_size,
                              hipStream_t stream) {
    const float* acts  = (const float*)d_in[0];
    const float* W     = (const float*)d_in[1];
    const float* bias  = (const float*)d_in[2];
    const int*   layer = (const int*)d_in[3];
    float* out = (float*)d_out;

    dim3 grid(D_DIM / BN, B_DIM / BM);   // 6 x 16 = 96 workgroups
    dim3 block(256);                     // 8 wave32s
    decoder_gemm_wmma<<<grid, block, SMEM_BYTES, stream>>>(acts, W, bias, layer, out);
}
